// MoETransformerBlock_69844758167945
// MI455X (gfx1250) — matrix operand order repaired and checked by reading, not compiled, not run on hardware
//
#include <hip/hip_runtime.h>
#include <hip/hip_bf16.h>
#include <stdint.h>

// ---------------------------------------------------------------------------
// MoE transformer block forward, MI455X (gfx1250, wave32, WMMA bf16 + TDM).
// Workspace layout (requires ws_size >= ~70 MB):
//   [0)            x_bf16   : 32768*512*2        = 33,554,432 B
//   [33554432)     W1TB     : 8*2048*512*2       = 16,777,216 B   [e][c][nf=64][k=512]
//   [50331648)     W2TB     : 8*2048*512*2       = 16,777,216 B   [e][c][n=512][kf=64]
//   [67108864)     cnt      : 8 * int (padded to 256 B)
//   [67109120)     toklist  : 8*32768 * int      =  1,048,576 B
//   [68157696)     wgtlist  : 8*32768 * float    =  1,048,576 B
// ---------------------------------------------------------------------------

#define T_TOKENS 32768
#define DM 512
#define DFF 2048
#define NE 8
#define FC 64                 // ff chunk size
#define NCHUNK (DFF / FC)     // 32

typedef __bf16 bf16_t;
typedef __attribute__((ext_vector_type(16))) __bf16 v16bf;
typedef __attribute__((ext_vector_type(8)))  float  v8f;
typedef uint32_t u32;
typedef __attribute__((ext_vector_type(4))) u32 u32x4;
typedef __attribute__((ext_vector_type(8))) u32 u32x8;

// LDS layout (dynamic). Row strides padded: 1072 B (=268 dwords, %64==12 ->
// conflict-free for 16-lane row stride) and 144 B (=36 dwords, gcd(36,64)=4).
// Double-buffered W chunks for TDM/compute overlap.
#define LDS_OX    0u                       // x tile [16][512+24] bf16, row 1072 B
#define LDS_OH    17152u                   // h tile [16][64+8]   bf16, row  144 B
#define LDS_OW1A  19456u                   // W1T chunk buf0 [64][512+24]
#define LDS_OW2A  88064u                   // W2T chunk buf0 [512][64+8]
#define LDS_OW1B  161792u                  // W1T chunk buf1
#define LDS_OW2B  230400u                  // W2T chunk buf1
#define LDS_SIZE  304128u                  // < 320 KB WGP LDS

union FragU { uint4 u[2]; v16bf v; };

// A fragment, 16x32 bf16: lanes0-15 K0..15,
// lanes16-31 K16..31; row M = lane&15. Two 16B LDS loads.
__device__ __forceinline__ v16bf frag_a(const char* base, int rowStride,
                                        int rbase, int kb, int lane) {
  int r = lane & 15, hi = lane >> 4;
  const char* p = base + (rbase + r) * rowStride + (kb + hi * 16) * 2;
  FragU f;
  f.u[0] = *(const uint4*)p;
  f.u[1] = *(const uint4*)(p + 16);   // +8 elements
  return f.v;
}

// B fragment from n-major (transposed) storage: lanes0-15 col=lane K0..15,
// lanes16-31 col=lane-16 K16..31. Two contiguous 16B LDS loads.
__device__ __forceinline__ v16bf frag_b(const char* base, int rowStride,
                                        int nbase, int kb, int lane) {
  int r = lane & 15, hi = lane >> 4;
  const char* p = base + (nbase + r) * rowStride + (kb + hi * 16) * 2;
  FragU f;
  f.u[0] = *(const uint4*)p;
  f.u[1] = *(const uint4*)(p + 16);
  return f.v;
}

// CDNA5 async global->LDS copy (16B per lane), ASYNCcnt-tracked.
__device__ __forceinline__ void async_cp16(unsigned lds_off, const void* gsrc) {
  asm volatile("global_load_async_to_lds_b128 %0, %1, off"
               :: "v"(lds_off), "v"(gsrc) : "memory");
}
__device__ __forceinline__ void wait_async0() {
  asm volatile("s_wait_asynccnt 0x0" ::: "memory");
}

// ---- Tensor Data Mover descriptors (CDNA5 ISA) -------
// Group0: [1:0]=count(1), [63:32]=lds_addr, [120:64]=global_addr, [127:126]=2
__device__ __forceinline__ u32x4 tdm_g0(unsigned lds_addr, const void* gaddr) {
  uint64_t ga = (uint64_t)gaddr;
  u32x4 g;
  g[0] = 1u;
  g[1] = lds_addr;
  g[2] = (u32)ga;
  g[3] = (u32)(ga >> 32) | (2u << 30);
  return g;
}
// Group1: data_size [17:16], pad_en [20], pad_interval [24:22], pad_amount
// [31:25], tensor_dim0 [79:48], tensor_dim1 [111:80], tile_dim0 [127:112],
// tile_dim1 [143:128], tensor_dim0_stride [207:160]. workgroup_mask=0.
__device__ __forceinline__ u32x8 tdm_g1(u32 dsz, u32 pad_int, u32 pad_amt,
                                        u32 td0, u32 td1, u32 tile0, u32 tile1,
                                        u32 stride0) {
  u32x8 g;
  g[0] = (dsz << 16) | (1u << 20) | (pad_int << 22) | (pad_amt << 25);
  g[1] = (td0 & 0xFFFFu) << 16;
  g[2] = (td0 >> 16) | ((td1 & 0xFFFFu) << 16);
  g[3] = (td1 >> 16) | (tile0 << 16);
  g[4] = tile1 & 0xFFFFu;
  g[5] = stride0;
  g[6] = 0u;
  g[7] = 0u;
  return g;
}
__device__ __forceinline__ void tdm_load_2d(u32x4 g0, u32x8 g1) {
  asm volatile("tensor_load_to_lds %0, %1" :: "s"(g0), "s"(g1) : "memory");
}

// ---------------------------------------------------------------------------
__global__ void zero_out_kernel(float* __restrict__ out, int n, int* __restrict__ cnt) {
  int i = blockIdx.x * blockDim.x + threadIdx.x;
  if (i < n) out[i] = 0.f;
  if (i < NE) cnt[i] = 0;
}

__global__ void cvt_x_kernel(const float* __restrict__ x, bf16_t* __restrict__ xb) {
  int i = blockIdx.x * blockDim.x + threadIdx.x;   // < T*DM
  xb[i] = (bf16_t)x[i];
}

// W1 [e][k=512][n=2048] fp32 -> W1TB [e][c=32][nf=64][k=512] bf16 (n=c*64+nf)
__global__ void cvt_w1_kernel(const float* __restrict__ w1, bf16_t* __restrict__ w1tb) {
  int i = blockIdx.x * blockDim.x + threadIdx.x;
  int k  = i & 511;
  int t1 = i >> 9;
  int nf = t1 & 63;
  int t2 = t1 >> 6;
  int c  = t2 & 31;
  int e  = t2 >> 5;
  int n  = c * 64 + nf;
  w1tb[i] = (bf16_t)w1[((size_t)e * DM + k) * DFF + n];
}

// W2 [e][k=2048][n=512] fp32 -> W2TB [e][c=32][n=512][kf=64] bf16 (k=c*64+kf)
__global__ void cvt_w2_kernel(const float* __restrict__ w2, bf16_t* __restrict__ w2tb) {
  int i = blockIdx.x * blockDim.x + threadIdx.x;
  int kf = i & 63;
  int t1 = i >> 6;
  int n  = t1 & 511;
  int t2 = t1 >> 9;
  int c  = t2 & 31;
  int e  = t2 >> 5;
  int k  = c * 64 + kf;
  w2tb[i] = (bf16_t)w2[((size_t)e * DFF + k) * DM + n];
}

// Router: logits = x @ Wg, top-2, softmax over the 2, append to expert lists.
__global__ void router_topk_kernel(const float* __restrict__ x,
                                   const float* __restrict__ wg,
                                   int* __restrict__ cnt,
                                   int* __restrict__ toklist,
                                   float* __restrict__ wgtlist) {
  int t = blockIdx.x * blockDim.x + threadIdx.x;
  if (t >= T_TOKENS) return;
  const float* xr = x + (size_t)t * DM;
  float acc[NE];
#pragma unroll
  for (int e = 0; e < NE; ++e) acc[e] = 0.f;
  for (int d = 0; d < DM; ++d) {
    float xv = xr[d];
#pragma unroll
    for (int e = 0; e < NE; ++e) acc[e] += xv * wg[d * NE + e];
  }
  int e1 = 0; float l1 = acc[0];
#pragma unroll
  for (int e = 1; e < NE; ++e) if (acc[e] > l1) { l1 = acc[e]; e1 = e; }
  int e2 = -1; float l2 = -3.4e38f;
#pragma unroll
  for (int e = 0; e < NE; ++e) if (e != e1 && acc[e] > l2) { l2 = acc[e]; e2 = e; }
  float wA = 1.f / (1.f + __expf(l2 - l1));
  float wB = 1.f - wA;
  int p1 = atomicAdd(&cnt[e1], 1);
  toklist[e1 * T_TOKENS + p1] = t; wgtlist[e1 * T_TOKENS + p1] = wA;
  int p2 = atomicAdd(&cnt[e2], 1);
  toklist[e2 * T_TOKENS + p2] = t; wgtlist[e2 * T_TOKENS + p2] = wB;
}

// ---------------------------------------------------------------------------
// Main routed expert GEMM. Block = (tile of 16 routed tokens) x (expert).
// 128 threads = 4 waves; TDM double-buffers W chunks while waves compute.
__global__ __launch_bounds__(128)
void moe_expert_gemm(const bf16_t* __restrict__ xb,
                     const bf16_t* __restrict__ w1tb,
                     const bf16_t* __restrict__ w2tb,
                     const int* __restrict__ cnt,
                     const int* __restrict__ toklist,
                     const float* __restrict__ wgtlist,
                     float* __restrict__ out) {
  extern __shared__ char smem[];
  const int e    = blockIdx.y;
  const int tile = blockIdx.x;
  const int n_e  = cnt[e];
  if (tile * 16 >= n_e) return;            // uniform early exit

  const int tid  = threadIdx.x;
  const int lane = tid & 31;
  const int wv   = tid >> 5;

  __shared__ int   s_tok[16];
  __shared__ float s_wgt[16];

  const unsigned smem0 = (unsigned)(size_t)smem;   // LDS byte offset of base
  const unsigned OW1[2] = {LDS_OW1A, LDS_OW1B};
  const unsigned OW2[2] = {LDS_OW2A, LDS_OW2B};

  if (tid < 16) {
    int idx = tile * 16 + tid;
    s_tok[tid] = (idx < n_e) ? toklist[e * T_TOKENS + idx] : 0;
    s_wgt[tid] = (idx < n_e) ? wgtlist[e * T_TOKENS + idx] : 0.f;
  }

  // TDM prefetch of chunk 0 into buffer 0 (wave 0 only; EXEC-independent DMA).
  if (wv == 0) {
    const char* w1src = (const char*)w1tb + (size_t)(e * 32 + 0) * (64 * 512 * 2);
    const char* w2src = (const char*)w2tb + (size_t)(e * 32 + 0) * (512 * 64 * 2);
    // W1 chunk: 64 rows x 512 bf16, pad 1024B rows (256 dw -> code 7) by 48B (12 dw -> code 11)
    tdm_load_2d(tdm_g0(smem0 + OW1[0], w1src),
                tdm_g1(1u, 7u, 11u, 512u, 64u, 512u, 64u, 512u));
    // W2 chunk: 512 rows x 64 bf16, pad 128B rows (32 dw -> code 4) by 16B (4 dw -> code 3)
    tdm_load_2d(tdm_g0(smem0 + OW2[0], w2src),
                tdm_g1(1u, 4u, 3u, 64u, 512u, 64u, 512u, 64u));
  }

  // Async gather of 16 token rows of x_bf16 (1 KB each) into LDS.
  for (int s = tid; s < 1024; s += 128) {
    int m = s >> 6, c16 = s & 63;
    int idx = tile * 16 + m;
    int tok = (idx < n_e) ? toklist[e * T_TOKENS + idx] : 0;
    const char* src = (const char*)xb + (size_t)tok * (DM * 2) + c16 * 16;
    async_cp16(smem0 + LDS_OX + m * 1072 + c16 * 16, src);
  }
  wait_async0();

  v8f acc[8];
#pragma unroll
  for (int j = 0; j < 8; ++j) acc[j] = (v8f){0.f,0.f,0.f,0.f,0.f,0.f,0.f,0.f};

  for (int c = 0; c < NCHUNK; ++c) {
    const int buf = c & 1;
    if (wv == 0) __builtin_amdgcn_s_wait_tensorcnt(0);   // chunk c resident
    __syncthreads();
    // Kick DMA for chunk c+1 into the other buffer; overlaps compute below.
    if (wv == 0 && (c + 1) < NCHUNK) {
      const char* w1src = (const char*)w1tb + (size_t)(e * 32 + c + 1) * (64 * 512 * 2);
      const char* w2src = (const char*)w2tb + (size_t)(e * 32 + c + 1) * (512 * 64 * 2);
      tdm_load_2d(tdm_g0(smem0 + OW1[buf ^ 1], w1src),
                  tdm_g1(1u, 7u, 11u, 512u, 64u, 512u, 64u, 512u));
      tdm_load_2d(tdm_g0(smem0 + OW2[buf ^ 1], w2src),
                  tdm_g1(1u, 4u, 3u, 64u, 512u, 64u, 512u, 64u));
    }

    const char* sw1 = smem + OW1[buf];
    const char* sw2 = smem + OW2[buf];

    // GEMM1: h[16][FC] — wave wv owns ff cols [wv*16, wv*16+16).
    // Two independent accumulator chains (even/odd K) to break the
    // WMMA->WMMA dependency and fill the 4-coexec hazard slots.
    v8f hc0 = (v8f){0.f,0.f,0.f,0.f,0.f,0.f,0.f,0.f};
    v8f hc1 = (v8f){0.f,0.f,0.f,0.f,0.f,0.f,0.f,0.f};
#pragma unroll 2
    for (int kb = 0; kb < DM; kb += 64) {
      v16bf a0 = frag_a(smem + LDS_OX, 1072, 0,       kb,      lane);
      v16bf b0 = frag_b(sw1,           1072, wv * 16, kb,      lane);
      v16bf a1 = frag_a(smem + LDS_OX, 1072, 0,       kb + 32, lane);
      v16bf b1 = frag_b(sw1,           1072, wv * 16, kb + 32, lane);
      hc0 = __builtin_amdgcn_wmma_f32_16x16x32_bf16(false, a0, false, b0,
                                                    (short)0, hc0, false, false);
      hc1 = __builtin_amdgcn_wmma_f32_16x16x32_bf16(false, a1, false, b1,
                                                    (short)0, hc1, false, false);
    }
    // SiLU (fast rcp, no IEEE divide) + bf16 + store h tile to LDS.
    {
      int hi = lane >> 4, nn = wv * 16 + (lane & 15);
      bf16_t* hrow = (bf16_t*)(smem + LDS_OH);
#pragma unroll
      for (int r = 0; r < 8; ++r) {
        int m = hi * 8 + r;
        float v = hc0[r] + hc1[r];
        float s = v * __builtin_amdgcn_rcpf(1.f + __expf(-v));
        hrow[m * 72 + nn] = (bf16_t)s;
      }
    }
    __syncthreads();

    // GEMM2: out[16][512] += h @ W2chunk — wave wv owns cols [wv*128,+128).
    // 8 independent accumulators per K-step: no WMMA->WMMA RAW chains.
#pragma unroll
    for (int kb = 0; kb < FC; kb += 32) {
      v16bf a = frag_a(smem + LDS_OH, 144, 0, kb, lane);
#pragma unroll
      for (int j = 0; j < 8; ++j) {
        v16bf b = frag_b(sw2, 144, wv * 128 + j * 16, kb, lane);
        acc[j] = __builtin_amdgcn_wmma_f32_16x16x32_bf16(false, a, false, b,
                                                         (short)0, acc[j], false, false);
      }
    }
    __syncthreads();   // buf reads complete before TDM overwrites next iter
  }

  // Gate-weighted scatter-add (HW f32 atomics; <=2 experts collide per token).
  int hi = lane >> 4, nl = lane & 15;
#pragma unroll
  for (int j = 0; j < 8; ++j) {
    int nb = wv * 128 + j * 16 + nl;
#pragma unroll
    for (int r = 0; r < 8; ++r) {
      int m = hi * 8 + r;
      unsafeAtomicAdd(&out[(size_t)s_tok[m] * DM + nb], s_wgt[m] * acc[j][r]);
    }
  }
}

// ---------------------------------------------------------------------------
extern "C" void kernel_launch(void* const* d_in, const int* in_sizes, int n_in,
                              void* d_out, int out_size, void* d_ws, size_t ws_size,
                              hipStream_t stream) {
  const float* x  = (const float*)d_in[0];   // [8,4096,512]
  const float* Wg = (const float*)d_in[1];   // [512,8]
  const float* W1 = (const float*)d_in[2];   // [8,512,2048]
  const float* W2 = (const float*)d_in[3];   // [8,2048,512]
  float* out = (float*)d_out;                // [8,4096,512]

  char* ws = (char*)d_ws;
  bf16_t* xb   = (bf16_t*)(ws);
  bf16_t* w1tb = (bf16_t*)(ws + 33554432);
  bf16_t* w2tb = (bf16_t*)(ws + 50331648);
  int*    cnt  = (int*)   (ws + 67108864);
  int*    tokl = (int*)   (ws + 67109120);
  float*  wgtl = (float*) (ws + 68157696);

  zero_out_kernel<<<65536, 256, 0, stream>>>(out, T_TOKENS * DM, cnt);
  cvt_x_kernel  <<<65536, 256, 0, stream>>>(x, xb);
  cvt_w1_kernel <<<32768, 256, 0, stream>>>(W1, w1tb);
  cvt_w2_kernel <<<32768, 256, 0, stream>>>(W2, w2tb);
  router_topk_kernel<<<T_TOKENS / 256, 256, 0, stream>>>(x, Wg, cnt, tokl, wgtl);

  dim3 grid(T_TOKENS / 16, NE);   // worst case: every token routed to expert e
  moe_expert_gemm<<<grid, 128, LDS_SIZE, stream>>>(xb, w1tb, w2tb, cnt, tokl, wgtl, out);
}
